// ScannedRNN_29300266893353
// MI455X (gfx1250) — compile-verified, matrix operand order checked by reading, not run on hardware
//
#include <hip/hip_runtime.h>

// GRU scan, persistent WMMA kernel for MI455X (gfx1250, wave32).
// T=512 steps, B=256, H=512. Math: bf16 WMMA (16x16x32) with f32 accumulate.
// Workspace layout (needs ~4 MB):
//   [0x000000) WiSw  : bf16, WMMA-swizzled [96 strips][16 kb][32 lanes][16]  (1.5 MB)
//   [0x180000) WhSw  : same                                                  (1.5 MB)
//   [0x300000) hA    : f32 [256][512] state buffer                           (512 KB)
//   [0x380000) hB    : f32 [256][512] state buffer                           (512 KB)
//   [0x400000) bar   : int[2] device barrier {count, generation}

#define T_STEPS 512
#define BATCH   256
#define HID     512
#define HID3    1536
#define NBLOCKS 64
#define NTHREADS 256

typedef __attribute__((ext_vector_type(16))) __bf16 v16bf;
typedef __attribute__((ext_vector_type(8)))  float  v8f;

__device__ __forceinline__ v8f wmma_bf16(v16bf a, v16bf b, v8f c) {
  // D = A(16x32 bf16) * B(32x16 bf16) + C(16x16 f32)
  return __builtin_amdgcn_wmma_f32_16x16x32_bf16(
      /*neg_a=*/false, a, /*neg_b=*/false, b,
      /*c_mod=*/(short)0, c, /*reuse_a=*/false, /*reuse_b=*/false);
}

// A fragment (16x32, bf16) from row-major f32 [rows x HID].
// Lane l holds row (l&15); lanes 0-15 K={k0..k0+7, k0+16..k0+23},
// lanes 16-31 K={k0+8..k0+15, k0+24..k0+31}. `scale` folds the reset mask.
__device__ __forceinline__ v16bf load_a_f32(const float* __restrict__ src,
                                            int row0, int k0, int lane, float scale) {
  const float* p = src + (size_t)(row0 + (lane & 15)) * HID + k0 + ((lane & 16) ? 8 : 0);
  const float4* q = (const float4*)p;
  float4 f0 = q[0], f1 = q[1], f2 = q[4], f3 = q[5];
  v16bf a;
  a[0]  = (__bf16)(f0.x * scale); a[1]  = (__bf16)(f0.y * scale);
  a[2]  = (__bf16)(f0.z * scale); a[3]  = (__bf16)(f0.w * scale);
  a[4]  = (__bf16)(f1.x * scale); a[5]  = (__bf16)(f1.y * scale);
  a[6]  = (__bf16)(f1.z * scale); a[7]  = (__bf16)(f1.w * scale);
  a[8]  = (__bf16)(f2.x * scale); a[9]  = (__bf16)(f2.y * scale);
  a[10] = (__bf16)(f2.z * scale); a[11] = (__bf16)(f2.w * scale);
  a[12] = (__bf16)(f3.x * scale); a[13] = (__bf16)(f3.y * scale);
  a[14] = (__bf16)(f3.z * scale); a[15] = (__bf16)(f3.w * scale);
  return a;
}

// B fragment (32x16, bf16) from pre-swizzled weights: one contiguous 32B per lane.
__device__ __forceinline__ v16bf load_b(const __bf16* __restrict__ Wsw,
                                        int strip, int kb, int lane) {
  return *(const v16bf*)(Wsw + ((((size_t)strip * 16 + kb) * 32 + lane) << 4));
}

// Device-wide sense-reversing barrier (blocks are guaranteed co-resident: 64 blocks).
__device__ __forceinline__ void grid_barrier(int* __restrict__ bar, int nblocks) {
  __threadfence();
  __syncthreads();
  if (threadIdx.x == 0) {
    volatile int* cnt = bar;
    volatile int* gen = bar + 1;
    int g = *gen;
    int old = atomicAdd(bar, 1);
    if (old == nblocks - 1) {
      *cnt = 0;
      __threadfence();
      atomicAdd(bar + 1, 1);
    } else {
      while (*gen == g) __builtin_amdgcn_s_sleep(4);
    }
    __threadfence();
  }
  __syncthreads();
}

// Convert Wi/Wh (f32 [HID x HID3] row-major) into WMMA-swizzled bf16, copy h0,
// and reset the barrier. Runs before the persistent kernel on the same stream.
__global__ void __launch_bounds__(256)
gru_prep(const float* __restrict__ Wi, const float* __restrict__ Wh,
         const float* __restrict__ h0,
         __bf16* __restrict__ WiSw, __bf16* __restrict__ WhSw,
         float* __restrict__ hA, int* __restrict__ bar) {
  int tid = blockIdx.x * blockDim.x + threadIdx.x;
  int stride = gridDim.x * blockDim.x;
  if (tid == 0) { bar[0] = 0; bar[1] = 0; }
  for (int i = tid; i < BATCH * HID; i += stride) hA[i] = h0[i];
  const int units = 96 * 16 * 32;  // strips * kb * lanes, per matrix
  for (int u = tid; u < 2 * units; u += stride) {
    int m = u / units;
    int r = u - m * units;
    int lane = r & 31;
    int kb = (r >> 5) & 15;
    int s = r >> 9;
    const float* W = m ? Wh : Wi;
    __bf16* D = (m ? WhSw : WiSw) + (size_t)r * 16;
    int ncol  = s * 16 + (lane & 15);
    int kbase = kb * 32 + ((lane & 16) ? 8 : 0);
#pragma unroll
    for (int j = 0; j < 8; ++j) {
      D[j]     = (__bf16)W[(size_t)(kbase + j)      * HID3 + ncol];
      D[j + 8] = (__bf16)W[(size_t)(kbase + j + 16) * HID3 + ncol];
    }
  }
}

__global__ void __launch_bounds__(NTHREADS)
gru_scan(const float* __restrict__ ins, const float* __restrict__ bi,
         const float* __restrict__ bhn, const unsigned char* __restrict__ resets,
         const __bf16* __restrict__ WiSw, const __bf16* __restrict__ WhSw,
         float* __restrict__ hA, float* __restrict__ hB,
         float* __restrict__ out, int* __restrict__ bar) {
  const int wave = threadIdx.x >> 5;
  const int lane = threadIdx.x & 31;
  const int tile = blockIdx.x * 8 + wave;   // 0..511: all 8 waves share tn
  const int tm = tile & 15;                 // batch tile (B/16 = 16)
  const int tn = tile >> 4;                 // H column tile (H/16 = 32)
  const int row0 = tm * 16;
  const int n = tn * 16 + (lane & 15);      // this lane's output column
  const float b_ir = bi[n], b_iz = bi[HID + n], b_in = bi[2 * HID + n];
  const float b_hn = bhn[n];
  float* hc  = hA;
  float* hnx = hB;

  for (int t = 0; t < T_STEPS; ++t) {
    const float* x = ins + (size_t)t * BATCH * HID;
    const unsigned char* rst = resets + (size_t)t * BATCH;
    const float am = rst[row0 + (lane & 15)] ? 0.f : 1.f;  // reset mask for A rows

    v8f ir = {}, iz = {}, in_ = {}, hr = {}, hz = {}, hn_ = {};
#pragma unroll 4
    for (int kb = 0; kb < 16; ++kb) {
      const int k0 = kb * 32;
      v16bf ax = load_a_f32(x,  row0, k0, lane, 1.0f);
      v16bf ah = load_a_f32(hc, row0, k0, lane, am);
      ir  = wmma_bf16(ax, load_b(WiSw, tn,      kb, lane), ir);
      iz  = wmma_bf16(ax, load_b(WiSw, tn + 32, kb, lane), iz);
      in_ = wmma_bf16(ax, load_b(WiSw, tn + 64, kb, lane), in_);
      hr  = wmma_bf16(ah, load_b(WhSw, tn,      kb, lane), hr);
      hz  = wmma_bf16(ah, load_b(WhSw, tn + 32, kb, lane), hz);
      hn_ = wmma_bf16(ah, load_b(WhSw, tn + 64, kb, lane), hn_);
    }

    // Epilogue: C/D layout => element e, lane l holds (M = e + 8*(l>>4), N = l&15)
    float* outp = out + (size_t)t * BATCH * HID;
#pragma unroll
    for (int e = 0; e < 8; ++e) {
      int m = row0 + e + ((lane >> 4) << 3);
      float hp = rst[m] ? 0.f : hc[(size_t)m * HID + n];
      float r  = 1.f / (1.f + __expf(-(ir[e] + b_ir + hr[e])));
      float z  = 1.f / (1.f + __expf(-(iz[e] + b_iz + hz[e])));
      float nv = in_[e] + b_in + r * (hn_[e] + b_hn);
      nv = fminf(fmaxf(nv, -15.f), 15.f);
      float e2 = __expf(2.f * nv);
      float ng = (e2 - 1.f) / (e2 + 1.f);        // tanh
      float hv = (1.f - z) * ng + z * hp;
      hnx[(size_t)m * HID + n]  = hv;
      outp[(size_t)m * HID + n] = hv;
    }

    if (t + 1 < T_STEPS) {  // warm L2 for next step's x rows (global_prefetch_b8)
      __builtin_prefetch(ins + (size_t)(t + 1) * BATCH * HID +
                             (size_t)(row0 + (lane & 15)) * HID + 4 * lane, 0, 0);
    }

    grid_barrier(bar, NBLOCKS);
    float* tmp = hc; hc = hnx; hnx = tmp;   // swap state buffers
  }
}

extern "C" void kernel_launch(void* const* d_in, const int* in_sizes, int n_in,
                              void* d_out, int out_size, void* d_ws, size_t ws_size,
                              hipStream_t stream) {
  // setup_inputs order: h0, ins, Wi, bi, Wh, bhn, resets
  const float* h0  = (const float*)d_in[0];
  const float* ins = (const float*)d_in[1];
  const float* Wi  = (const float*)d_in[2];
  const float* bi  = (const float*)d_in[3];
  const float* Wh  = (const float*)d_in[4];
  const float* bhn = (const float*)d_in[5];
  const unsigned char* resets = (const unsigned char*)d_in[6];  // jnp.bool_ = 1 byte

  char* ws = (char*)d_ws;                       // need ~4 MB
  __bf16* WiSw = (__bf16*)(ws + 0x000000);
  __bf16* WhSw = (__bf16*)(ws + 0x180000);
  float*  hA   = (float*)(ws + 0x300000);
  float*  hB   = (float*)(ws + 0x380000);
  int*    bar  = (int*)  (ws + 0x400000);

  gru_prep<<<512, 256, 0, stream>>>(Wi, Wh, h0, WiSw, WhSw, hA, bar);
  gru_scan<<<NBLOCKS, NTHREADS, 0, stream>>>(ins, bi, bhn, resets,
                                             WiSw, WhSw, hA, hB,
                                             (float*)d_out, bar);
}
